// Transformer_17351667876507
// MI455X (gfx1250) — matrix operand order repaired and checked by reading, not compiled, not run on hardware
//
#include <hip/hip_runtime.h>
#include <hip/hip_bf16.h>
#include <stdint.h>

// ---------------- model constants (match reference) ----------------
#define DMODEL 512
#define NHEAD  8
#define NLAYER 6
#define DFFN   2048
#define DKH    64
#define BB     4
#define SSRC   1024
#define TTGT   1024
#define VTGT   32000
#define EPS_LN 1e-5f

typedef __attribute__((ext_vector_type(16))) __bf16 v16bf;
typedef __attribute__((ext_vector_type(8)))  float  v8f;
typedef __hip_bfloat16 bf16;

// ==================================================================
// Generic batched bf16 GEMM with WMMA:  C[z] = A[z] * B[z] (+bias)(relu)
// Block tile 256(M) x 64(N), K-step 32, double-buffered LDS with an
// explicit software pipeline:
//    nx = global loads (k+32)  ->  8 WMMAs on live buffer  ->
//    LDS stores of nx          ->  barrier
// so the global_load_b128s stay in flight behind the WMMA block.
// 8 waves; each wave owns two 16-row strips and 4 N-sub-tiles.
// All M multiples of 256, N multiples of 64, K multiples of 32.
// ==================================================================
struct GemmArgs {
  const unsigned short* A; int lda; long long aOuter, aInner;
  const unsigned short* B; int ldb; long long bOuter, bInner;
  void* C; int ldc; long long cOuter, cInner;
  const float* bias;
  int K;
  int innerCount;   // z = bo*innerCount + bi
  int transB;       // B stored [N,K] instead of [K,N]
  int relu;
  int outBf16;
};

struct Staged {
  uint4 a0, a1, a2, a3;        // A-tile row (32 bf16)
  uint4 bv;                    // B-tile chunk (transB path)
  unsigned short bs[8];        // B-tile gather (non-trans path)
};

__global__ __launch_bounds__(256) void gemm_bf16_wmma(GemmArgs g) {
  __shared__ alignas(16) unsigned short ldsA [2][256 * 40]; // [row][k], pad 40
  __shared__ alignas(16) unsigned short ldsBt[2][ 64 * 40]; // [n][k],   pad 40

  const int tid  = threadIdx.x;
  const int wave = tid >> 5;
  const int lane = tid & 31;
  const int laneM  = lane & 15;
  const int laneHi = lane >> 4;

  const int z  = blockIdx.z;
  const int bo = z / g.innerCount;
  const int bi = z % g.innerCount;

  const unsigned short* Ag = g.A + (long long)bo * g.aOuter + (long long)bi * g.aInner;
  const unsigned short* Bg = g.B + (long long)bo * g.bOuter + (long long)bi * g.bInner;
  const long long cbase = (long long)bo * g.cOuter + (long long)bi * g.cInner;

  const int row0 = blockIdx.y * 256;
  const int col0 = blockIdx.x * 64;

  v8f acc[2][4];
#pragma unroll
  for (int s = 0; s < 2; ++s)
#pragma unroll
    for (int ns = 0; ns < 4; ++ns)
#pragma unroll
      for (int r = 0; r < 8; ++r) acc[s][ns][r] = 0.0f;

  // ---- pipeline stage 1: issue global loads for tile at k0 ----
  auto loadGlobal = [&](int k0) -> Staged {
    Staged st;
    const uint4* srcA = (const uint4*)(Ag + (long long)(row0 + tid) * g.lda + k0);
    st.a0 = srcA[0]; st.a1 = srcA[1]; st.a2 = srcA[2]; st.a3 = srcA[3];
    if (g.transB) {
      const int n = tid >> 2;
      const int q = tid & 3;
      st.bv = *(const uint4*)(Bg + (long long)(col0 + n) * g.ldb + k0 + q * 8);
    } else {
      const int n  = tid & 63;
      const int kq = (tid >> 6) * 8;
      const unsigned short* src = Bg + (long long)(k0 + kq) * g.ldb + col0 + n;
#pragma unroll
      for (int kk = 0; kk < 8; ++kk) st.bs[kk] = src[(long long)kk * g.ldb];
    }
    return st;
  };
  // ---- pipeline stage 2: commit staged tile to LDS buffer ----
  auto storeLds = [&](const Staged& st, int buf) {
    uint4* dstA = (uint4*)(&ldsA[buf][tid * 40]);
    dstA[0] = st.a0; dstA[1] = st.a1; dstA[2] = st.a2; dstA[3] = st.a3;
    if (g.transB) {
      const int n = tid >> 2;
      const int q = tid & 3;
      *(uint4*)(&ldsBt[buf][n * 40 + q * 8]) = st.bv;
    } else {
      const int n  = tid & 63;
      const int kq = (tid >> 6) * 8;
      unsigned short* dst = &ldsBt[buf][n * 40 + kq];
#pragma unroll
      for (int kk = 0; kk < 8; ++kk) dst[kk] = st.bs[kk];
    }
  };

  {
    Staged st0 = loadGlobal(0);
    storeLds(st0, 0);
  }
  __syncthreads();

  int buf = 0;
  for (int k0 = 0; k0 < g.K; k0 += 32) {
    const bool hasNext = (k0 + 32 < g.K);
    Staged nx;
    if (hasNext) nx = loadGlobal(k0 + 32);   // loads issue here, consumed after WMMAs

    // ---- A fragments for both strips:
    //      lanes 0-15 -> K0-7 / K16-23 ; lanes 16-31 -> K8-15 / K24-31
    union AF { uint4 u[2]; v16bf v; } af0, af1;
    const unsigned short* la = &ldsA[buf][0];
    const int arow0 = (wave * 16 + laneM) * 40;
    const int arow1 = (128 + wave * 16 + laneM) * 40;
    af0.u[0] = *(const uint4*)(la + arow0 + laneHi * 8);
    af0.u[1] = *(const uint4*)(la + arow0 + 16 + laneHi * 8);
    af1.u[0] = *(const uint4*)(la + arow1 + laneHi * 8);
    af1.u[1] = *(const uint4*)(la + arow1 + 16 + laneHi * 8);

    const unsigned short* lb = &ldsBt[buf][0];
#pragma unroll
    for (int ns = 0; ns < 4; ++ns) {
      // B fragment: lane = N, slots j<8 -> K = j + 8*laneHi, j>=8 -> K = 16 + 8*laneHi + (j-8)
      union BF { uint4 u[2]; v16bf v; } bfrag;
      const int brow = (ns * 16 + laneM) * 40;
      bfrag.u[0] = *(const uint4*)(lb + brow + laneHi * 8);
      bfrag.u[1] = *(const uint4*)(lb + brow + 16 + laneHi * 8);

      acc[0][ns] = __builtin_amdgcn_wmma_f32_16x16x32_bf16(
          false, af0.v, false, bfrag.v, (short)0, acc[0][ns], false, false);
      acc[1][ns] = __builtin_amdgcn_wmma_f32_16x16x32_bf16(
          false, af1.v, false, bfrag.v, (short)0, acc[1][ns], false, false);
    }

    if (hasNext) storeLds(nx, buf ^ 1);      // single wait, then ds stores
    __syncthreads();
    buf ^= 1;
  }

  // ---- epilogue: bias / relu / write fp32 or bf16 ----
#pragma unroll
  for (int s = 0; s < 2; ++s) {
#pragma unroll
    for (int ns = 0; ns < 4; ++ns) {
      const int gc = col0 + ns * 16 + laneM;
      const float bv = g.bias ? g.bias[gc] : 0.0f;
#pragma unroll
      for (int r = 0; r < 8; ++r) {
        const int gr = row0 + s * 128 + wave * 16 + r + laneHi * 8;
        float v = acc[s][ns][r] + bv;
        if (g.relu) v = v > 0.0f ? v : 0.0f;
        const long long idx = cbase + (long long)gr * g.ldc + gc;
        if (g.outBf16) ((bf16*)g.C)[idx] = __float2bfloat16(v);
        else           ((float*)g.C)[idx] = v;
      }
    }
  }
}

// ==================================================================
// Embedding lookup + sinusoidal positional encoding
// ==================================================================
__global__ void embed_pe_kernel(const float* __restrict__ emb,
                                const int* __restrict__ tok,
                                float* __restrict__ out,
                                int seqLen, int d_model, long long n) {
  long long idx = (long long)blockIdx.x * blockDim.x + threadIdx.x;
  if (idx >= n) return;
  const int d = (int)(idx % d_model);
  const long long bs = idx / d_model;
  const int pos = (int)(bs % seqLen);
  const int t = tok[bs];
  const float freq = expf((float)(d & ~1) * (-9.210340371976184f / (float)d_model));
  const float ang = (float)pos * freq;
  const float pe = (d & 1) ? cosf(ang) : sinf(ang);
  out[idx] = emb[(long long)t * d_model + d] + pe;
}

// ==================================================================
// LayerNorm (optional fused residual):  out = LN(x (+ res)) * g + b
// 128 threads per row, D = 512
// ==================================================================
__global__ __launch_bounds__(128) void ln_kernel(const float* __restrict__ x,
                                                 const float* __restrict__ res,
                                                 const float* __restrict__ gamma,
                                                 const float* __restrict__ beta,
                                                 float* __restrict__ out) {
  __shared__ float red[128];
  const int row = blockIdx.x;
  const float* xr = x + (long long)row * DMODEL;
  const float* rr = res ? res + (long long)row * DMODEL : nullptr;
  float local[4];
  float s = 0.0f;
#pragma unroll
  for (int i = 0; i < 4; ++i) {
    const int d = threadIdx.x + i * 128;
    float v = xr[d] + (rr ? rr[d] : 0.0f);
    local[i] = v;
    s += v;
  }
  red[threadIdx.x] = s; __syncthreads();
  for (int off = 64; off > 0; off >>= 1) {
    if (threadIdx.x < off) red[threadIdx.x] += red[threadIdx.x + off];
    __syncthreads();
  }
  const float mean = red[0] / (float)DMODEL;
  __syncthreads();
  float s2 = 0.0f;
#pragma unroll
  for (int i = 0; i < 4; ++i) { float d = local[i] - mean; s2 += d * d; }
  red[threadIdx.x] = s2; __syncthreads();
  for (int off = 64; off > 0; off >>= 1) {
    if (threadIdx.x < off) red[threadIdx.x] += red[threadIdx.x + off];
    __syncthreads();
  }
  const float inv = rsqrtf(red[0] / (float)DMODEL + EPS_LN);
#pragma unroll
  for (int i = 0; i < 4; ++i) {
    const int d = threadIdx.x + i * 128;
    out[(long long)row * DMODEL + d] = (local[i] - mean) * inv * gamma[d] + beta[d];
  }
}

// ==================================================================
// Masked scaled softmax over last dim; writes bf16 probabilities.
// mode 0: mask[b*Sk + k] (src key mask) ; mode 1: mask[q*Sk + k] (causal)
// Sk == 1024, 256 threads/row -> 4 cached elements per thread.
// ==================================================================
__global__ __launch_bounds__(256) void softmax_kernel(const float* __restrict__ scores,
                                                      bf16* __restrict__ probs,
                                                      const int* __restrict__ mask,
                                                      int mode, int H, int Sq, int Sk,
                                                      float scale) {
  __shared__ float red[256];
  const int row = blockIdx.x;            // b*H*Sq + h*Sq + q
  const int q = row % Sq;
  const int b = row / (H * Sq);
  const float* sr = scores + (long long)row * Sk;
  bf16* pr = probs + (long long)row * Sk;
  const int tid = threadIdx.x;

  float lv[4];
  float vmax = -3.4e38f;
#pragma unroll
  for (int it = 0; it < 4; ++it) {
    const int k = tid + it * 256;
    const int mv = (mode == 0) ? mask[(long long)b * Sk + k]
                               : mask[(long long)q * Sk + k];
    const float v = mv ? sr[k] * scale : -1e9f;
    lv[it] = v;
    vmax = fmaxf(vmax, v);
  }
  red[tid] = vmax; __syncthreads();
  for (int off = 128; off > 0; off >>= 1) {
    if (tid < off) red[tid] = fmaxf(red[tid], red[tid + off]);
    __syncthreads();
  }
  const float m = red[0];
  __syncthreads();
  float sum = 0.0f;
#pragma unroll
  for (int it = 0; it < 4; ++it) { lv[it] = expf(lv[it] - m); sum += lv[it]; }
  red[tid] = sum; __syncthreads();
  for (int off = 128; off > 0; off >>= 1) {
    if (tid < off) red[tid] += red[tid + off];
    __syncthreads();
  }
  const float rinv = 1.0f / red[0];
#pragma unroll
  for (int it = 0; it < 4; ++it) {
    pr[tid + it * 256] = __float2bfloat16(lv[it] * rinv);
  }
}

// ==================================================================
// fp32 -> bf16 conversion
// ==================================================================
__global__ void f2bf_kernel(const float* __restrict__ in, bf16* __restrict__ out, long long n) {
  long long i = (long long)blockIdx.x * blockDim.x + threadIdx.x;
  const long long stride = (long long)gridDim.x * blockDim.x;
  for (; i < n; i += stride) out[i] = __float2bfloat16(in[i]);
}

// ==================================================================
// Host orchestration
// ==================================================================
extern "C" void kernel_launch(void* const* d_in, const int* in_sizes, int n_in,
                              void* d_out, int out_size, void* d_ws, size_t ws_size,
                              hipStream_t stream) {
  const int D = DMODEL, H = NHEAD, NL = NLAYER, DFF = DFFN, dk = DKH;
  const int B = BB, S = SSRC, T = TTGT, VT = VTGT;

  // -------- inputs (setup_inputs order) --------
  const float* src_emb    = (const float*)d_in[0];
  const float* tgt_emb    = (const float*)d_in[1];
  const float* enc_attn_w = (const float*)d_in[2];
  const float* enc_attn_b = (const float*)d_in[3];
  const float* enc_ffn_w1 = (const float*)d_in[4];
  const float* enc_ffn_b1 = (const float*)d_in[5];
  const float* enc_ffn_w2 = (const float*)d_in[6];
  const float* enc_ffn_b2 = (const float*)d_in[7];
  const float* enc_ln_g   = (const float*)d_in[8];
  const float* enc_ln_b   = (const float*)d_in[9];
  const float* enc_fg     = (const float*)d_in[10];
  const float* enc_fb     = (const float*)d_in[11];
  const float* dec_self_w = (const float*)d_in[12];
  const float* dec_self_b = (const float*)d_in[13];
  const float* dec_src_w  = (const float*)d_in[14];
  const float* dec_src_b  = (const float*)d_in[15];
  const float* dec_ffn_w1 = (const float*)d_in[16];
  const float* dec_ffn_b1 = (const float*)d_in[17];
  const float* dec_ffn_w2 = (const float*)d_in[18];
  const float* dec_ffn_b2 = (const float*)d_in[19];
  const float* dec_ln_g   = (const float*)d_in[20];
  const float* dec_ln_b   = (const float*)d_in[21];
  const float* dec_fg     = (const float*)d_in[22];
  const float* dec_fb     = (const float*)d_in[23];
  const float* gen_w      = (const float*)d_in[24];
  const float* gen_b      = (const float*)d_in[25];
  const int*   src        = (const int*)d_in[26];
  const int*   tgt        = (const int*)d_in[27];
  const int*   src_mask   = (const int*)d_in[28];
  const int*   tgt_mask   = (const int*)d_in[29];
  float* logits = (float*)d_out;
  (void)in_sizes; (void)n_in; (void)out_size; (void)ws_size;

  // -------- workspace carve-out --------
  size_t cur = 0;
  auto alloc = [&](size_t bytes) -> void* {
    cur = (cur + 255) & ~(size_t)255;
    void* p = (char*)d_ws + cur;
    cur += bytes;
    return p;
  };
  const long long nAttnW = (long long)NL * 4 * D * D;   // 6.29M
  const long long nFfn   = (long long)NL * D * DFF;     // 6.29M
  const long long nGenW  = (long long)D * VT;           // 16.38M
  const long long nAct   = (long long)B * S * D;        // 2.10M
  const long long nScore = (long long)B * H * S * S;    // 33.55M
  const long long nHid   = (long long)B * S * DFF;      // 8.39M

  bf16* wb_enc_attn = (bf16*)alloc(nAttnW * 2);
  bf16* wb_dec_self = (bf16*)alloc(nAttnW * 2);
  bf16* wb_dec_src  = (bf16*)alloc(nAttnW * 2);
  bf16* wb_enc_f1   = (bf16*)alloc(nFfn * 2);
  bf16* wb_enc_f2   = (bf16*)alloc(nFfn * 2);
  bf16* wb_dec_f1   = (bf16*)alloc(nFfn * 2);
  bf16* wb_dec_f2   = (bf16*)alloc(nFfn * 2);
  bf16* wb_gen      = (bf16*)alloc(nGenW * 2);

  float* x_f32   = (float*)alloc(nAct * 4);
  float* y_f32   = (float*)alloc(nAct * 4);
  float* mem_f32 = (float*)alloc(nAct * 4);
  float* aux_f32 = (float*)alloc(nAct * 4);
  bf16*  xb      = (bf16*)alloc(nAct * 2);
  bf16*  memb    = (bf16*)alloc(nAct * 2);
  bf16*  qb      = (bf16*)alloc(nAct * 2);
  bf16*  kb      = (bf16*)alloc(nAct * 2);
  bf16*  vb      = (bf16*)alloc(nAct * 2);
  bf16*  attnb   = (bf16*)alloc(nAct * 2);
  bf16*  hb      = (bf16*)alloc(nHid * 2);
  float* scores  = (float*)alloc(nScore * 4);
  bf16*  probs   = (bf16*)alloc(nScore * 2);

  // -------- helpers --------
  auto conv = [&](const float* in, bf16* out, long long n) {
    int blocks = (int)((n + 1023) / 1024);
    if (blocks > 65535) blocks = 65535;
    hipLaunchKernelGGL(f2bf_kernel, dim3(blocks), dim3(1024), 0, stream, in, out, n);
  };
  auto gemm = [&](const void* A, int lda, long long aO, long long aI,
                  const void* Bm, int ldb, long long bO, long long bI,
                  void* C, int ldc, long long cO, long long cI,
                  const float* bias, int M, int N, int K,
                  int batches, int innerCount, int transB, int relu, int outBf16) {
    GemmArgs ga;
    ga.A = (const unsigned short*)A; ga.lda = lda; ga.aOuter = aO; ga.aInner = aI;
    ga.B = (const unsigned short*)Bm; ga.ldb = ldb; ga.bOuter = bO; ga.bInner = bI;
    ga.C = C; ga.ldc = ldc; ga.cOuter = cO; ga.cInner = cI;
    ga.bias = bias; ga.K = K; ga.innerCount = innerCount;
    ga.transB = transB; ga.relu = relu; ga.outBf16 = outBf16;
    dim3 grid(N / 64, M / 256, batches);
    hipLaunchKernelGGL(gemm_bf16_wmma, grid, dim3(256), 0, stream, ga);
  };
  auto lnorm = [&](const float* x, const float* res, const float* g, const float* bta,
                   float* out, int rows) {
    hipLaunchKernelGGL(ln_kernel, dim3(rows), dim3(128), 0, stream, x, res, g, bta, out);
  };
  auto smax = [&](const float* sc, bf16* pb, const int* mask, int mode, int Sq, int Sk) {
    hipLaunchKernelGGL(softmax_kernel, dim3(B * H * Sq), dim3(256), 0, stream,
                       sc, pb, mask, mode, H, Sq, Sk, 0.125f /* 1/sqrt(64) */);
  };

  // Generic MHA block: q from Xq(bf16,[B,Sq,D]); k,v from Xkv(bf16,[B,Skv,D]).
  auto mha = [&](const bf16* Xq, int Sq, const bf16* Xkv, int Skv,
                 const bf16* W, const float* Bv, const int* mask, int maskMode,
                 float* out /* [B,Sq,D] fp32 */) {
    const int M = B * Sq, Mk = B * Skv;
    // projections (bf16 out)
    gemm(Xq,  D, 0, 0, W + 0 * (long long)D * D, D, 0, 0, qb, D, 0, 0,
         Bv + 0 * D, M, D, D, 1, 1, 0, 0, 1);
    gemm(Xkv, D, 0, 0, W + 1 * (long long)D * D, D, 0, 0, kb, D, 0, 0,
         Bv + 1 * D, Mk, D, D, 1, 1, 0, 0, 1);
    gemm(Xkv, D, 0, 0, W + 2 * (long long)D * D, D, 0, 0, vb, D, 0, 0,
         Bv + 2 * D, Mk, D, D, 1, 1, 0, 0, 1);
    // scores[b,h] = Q[b,h] (Sq x dk) @ K[b,h]^T   (batched, B*H)
    gemm(qb, D, (long long)Sq * D, dk,
         kb, D, (long long)Skv * D, dk,
         scores, Skv, (long long)H * Sq * Skv, (long long)Sq * Skv,
         nullptr, Sq, Skv, dk, B * H, H, /*transB=*/1, 0, 0);
    smax(scores, probs, mask, maskMode, Sq, Skv);
    // O[b,h] = P (Sq x Skv) @ V[b,h] (Skv x dk) -> attnb [B,Sq,H,dk]
    gemm(probs, Skv, (long long)H * Sq * Skv, (long long)Sq * Skv,
         vb, D, (long long)Skv * D, dk,
         attnb, D, (long long)Sq * D, dk,
         nullptr, Sq, dk, Skv, B * H, H, 0, 0, 1);
    // output projection (fp32 out)
    gemm(attnb, D, 0, 0, W + 3 * (long long)D * D, D, 0, 0, out, D, 0, 0,
         Bv + 3 * D, M, D, D, 1, 1, 0, 0, 0);
  };

  // -------- convert all weights to bf16 --------
  conv(enc_attn_w, wb_enc_attn, nAttnW);
  conv(dec_self_w, wb_dec_self, nAttnW);
  conv(dec_src_w,  wb_dec_src,  nAttnW);
  conv(enc_ffn_w1, wb_enc_f1,   nFfn);
  conv(enc_ffn_w2, wb_enc_f2,   nFfn);
  conv(dec_ffn_w1, wb_dec_f1,   nFfn);
  conv(dec_ffn_w2, wb_dec_f2,   nFfn);
  conv(gen_w,      wb_gen,      nGenW);

  // ==================== ENCODER ====================
  hipLaunchKernelGGL(embed_pe_kernel, dim3((int)(nAct / 256)), dim3(256), 0, stream,
                     src_emb, src, x_f32, S, D, nAct);
  for (int i = 0; i < NL; ++i) {
    const bf16*  W  = wb_enc_attn + (long long)i * 4 * D * D;
    const float* Bv = enc_attn_b + (long long)i * 4 * D;
    conv(x_f32, xb, nAct);
    mha(xb, S, xb, S, W, Bv, src_mask, /*mode=*/0, aux_f32);
    lnorm(x_f32, aux_f32, enc_ln_g + (long long)i * 2 * D, enc_ln_b + (long long)i * 2 * D,
          x_f32, B * S);
    // FFN
    conv(x_f32, xb, nAct);
    gemm(xb, D, 0, 0, wb_enc_f1 + (long long)i * D * DFF, DFF, 0, 0, hb, DFF, 0, 0,
         enc_ffn_b1 + (long long)i * DFF, B * S, DFF, D, 1, 1, 0, /*relu=*/1, 1);
    gemm(hb, DFF, 0, 0, wb_enc_f2 + (long long)i * DFF * D, D, 0, 0, aux_f32, D, 0, 0,
         enc_ffn_b2 + (long long)i * D, B * S, D, DFF, 1, 1, 0, 0, 0);
    lnorm(x_f32, aux_f32, enc_ln_g + (long long)i * 2 * D + D, enc_ln_b + (long long)i * 2 * D + D,
          x_f32, B * S);
  }
  lnorm(x_f32, nullptr, enc_fg, enc_fb, mem_f32, B * S);
  conv(mem_f32, memb, nAct);

  // ==================== DECODER ====================
  hipLaunchKernelGGL(embed_pe_kernel, dim3((int)(nAct / 256)), dim3(256), 0, stream,
                     tgt_emb, tgt, y_f32, T, D, nAct);
  for (int i = 0; i < NL; ++i) {
    // self-attention (causal)
    conv(y_f32, xb, nAct);
    mha(xb, T, xb, T, wb_dec_self + (long long)i * 4 * D * D,
        dec_self_b + (long long)i * 4 * D, tgt_mask, /*mode=*/1, aux_f32);
    lnorm(y_f32, aux_f32, dec_ln_g + (long long)i * 3 * D, dec_ln_b + (long long)i * 3 * D,
          y_f32, B * T);
    // cross-attention (memory K/V)
    conv(y_f32, xb, nAct);
    mha(xb, T, memb, S, wb_dec_src + (long long)i * 4 * D * D,
        dec_src_b + (long long)i * 4 * D, src_mask, /*mode=*/0, aux_f32);
    lnorm(y_f32, aux_f32, dec_ln_g + (long long)i * 3 * D + D, dec_ln_b + (long long)i * 3 * D + D,
          y_f32, B * T);
    // FFN
    conv(y_f32, xb, nAct);
    gemm(xb, D, 0, 0, wb_dec_f1 + (long long)i * D * DFF, DFF, 0, 0, hb, DFF, 0, 0,
         dec_ffn_b1 + (long long)i * DFF, B * T, DFF, D, 1, 1, 0, /*relu=*/1, 1);
    gemm(hb, DFF, 0, 0, wb_dec_f2 + (long long)i * DFF * D, D, 0, 0, aux_f32, D, 0, 0,
         dec_ffn_b2 + (long long)i * D, B * T, D, DFF, 1, 1, 0, 0, 0);
    lnorm(y_f32, aux_f32, dec_ln_g + (long long)i * 3 * D + 2 * D, dec_ln_b + (long long)i * 3 * D + 2 * D,
          y_f32, B * T);
  }
  lnorm(y_f32, nullptr, dec_fg, dec_fb, y_f32, B * T);

  // ==================== GENERATOR ====================
  conv(y_f32, xb, nAct);
  gemm(xb, D, 0, 0, wb_gen, VT, 0, 0, logits, VT, 0, 0,
       gen_b, B * T, VT, D, 1, 1, 0, 0, 0);
}
